// Model_GMC_57681410786037
// MI455X (gfx1250) — matrix operand order repaired and checked by reading, not compiled, not run on hardware
//
#include <hip/hip_runtime.h>
#include <hip/hip_bf16.h>
#include <stddef.h>

typedef __attribute__((ext_vector_type(16))) _Float16 v16h;
typedef __attribute__((ext_vector_type(8)))  _Float16 v8h;
typedef __attribute__((ext_vector_type(8)))  float    v8f;

#define LKW 11
#define KCW 24
#define EPSV 1e-5f
#define LDS_STRIDE 40   // halfs per row: 80B = 16B-aligned, bank-rotating

// ---------------------------------------------------------------------------
// Embedding lookup: h[i, 0:16] = emb[x[i]]
// ---------------------------------------------------------------------------
__global__ void k_embed(const int* __restrict__ x, const float* __restrict__ emb,
                        float* __restrict__ h, int n) {
  int i = blockIdx.x * blockDim.x + threadIdx.x;
  if (i < n * 16) {
    int row = i >> 4, c = i & 15;
    h[i] = emb[x[row] * 16 + c];
  }
}

// ---------------------------------------------------------------------------
// Fused neighbor-window message pass (one workgroup per node).
// ---------------------------------------------------------------------------
__global__ void k_msg(const float* __restrict__ hin,
                      const float* __restrict__ pos,
                      const int*   __restrict__ seq,
                      const float* __restrict__ ori,
                      const int*   __restrict__ batch,
                      const float* __restrict__ wk1, const float* __restrict__ bk1,
                      const float* __restrict__ wk2, const float* __restrict__ bk2,
                      float* __restrict__ hpre, int n, int ic, float r)
{
  const int i   = blockIdx.x;
  const int tid = threadIdx.x;
  __shared__ float sg[LKW][6];
  __shared__ float sa[LKW][KCW];
  __shared__ float sval[LKW];
  __shared__ int   sidx[LKW];
  __shared__ float scnt;

  if (tid < LKW) {
    int jj  = i - (LKW / 2) + tid;
    int idx = jj < 0 ? 0 : (jj > n - 1 ? n - 1 : jj);
    bool valid = (jj >= 0) && (jj < n) && (batch[idx] == batch[i]);
    float dx = pos[idx * 3 + 0] - pos[i * 3 + 0];
    float dy = pos[idx * 3 + 1] - pos[i * 3 + 1];
    float dz = pos[idx * 3 + 2] - pos[i * 3 + 2];
    float d2 = dx * dx + dy * dy + dz * dz;
    valid = valid && (d2 <= r * r);
    float ds = (float)(seq[idx] - seq[i]) * (1.0f / (float)LKW);
    float ox = ori[i * 3 + 0],  oy = ori[i * 3 + 1],  oz = ori[i * 3 + 2];
    float jx = ori[idx * 3 + 0], jy = ori[idx * 3 + 1], jz = ori[idx * 3 + 2];
    float oo = ox * jx + oy * jy + oz * jz;
    float od = (ox * dx + oy * dy + oz * dz) / r;
    sg[tid][0] = dx / r; sg[tid][1] = dy / r; sg[tid][2] = dz / r;
    sg[tid][3] = ds;     sg[tid][4] = oo;     sg[tid][5] = od;
    sidx[tid] = idx;
    sval[tid] = valid ? 1.0f : 0.0f;
  }
  __syncthreads();
  if (tid == 0) {
    float c = 0.0f;
    for (int j = 0; j < LKW; ++j) c += sval[j];
    scnt = fmaxf(c, 1.0f);
  }
  for (int t = tid; t < LKW * KCW; t += blockDim.x) {
    int j = t / KCW, k = t - j * KCW;
    float s = bk1[k];
    #pragma unroll
    for (int q = 0; q < 6; ++q) s += sg[j][q] * wk1[q * KCW + k];
    sa[j][k] = fmaxf(s, 0.0f);
  }
  __syncthreads();
  float invc = 1.0f / scnt;
  for (int c = tid; c < ic; c += blockDim.x) {
    float acc = 0.0f;
    for (int j = 0; j < LKW; ++j) {
      if (sval[j] > 0.0f) {
        float w = bk2[c];
        #pragma unroll
        for (int k = 0; k < KCW; ++k) w += sa[j][k] * wk2[k * ic + c];
        acc += w * hin[(size_t)sidx[j] * ic + c];
      }
    }
    hpre[(size_t)i * ic + c] = acc * invc;
  }
}

// ---------------------------------------------------------------------------
// WMMA GEMM, LDS-staged, 128x128 tile, 256 threads (8 waves).
// Per K-step (32): cooperative float4 loads convert f32 panels to f16 in LDS
// (B stored transposed with bank-rotating stride); each wave then issues
// 8 v_wmma_f32_16x16x32_f16 on its 16-row strip, sharing B fragments.
// MGUARD=false instantiation (all bulk GEMMs, M%128==0) has a branch-free
// steady-state loop; the K%32 tail (only K=16 here) runs once, guarded.
// ---------------------------------------------------------------------------
template <bool MGUARD>
__global__ void k_gemm_wmma(const float* __restrict__ A, const float* __restrict__ B,
                            float* __restrict__ C, int M, int K, int N, int relu)
{
  __shared__ __align__(16) _Float16 sA[128 * LDS_STRIDE];
  __shared__ __align__(16) _Float16 sB[128 * LDS_STRIDE];
  const int tid  = threadIdx.x;            // 0..255
  const int row0 = blockIdx.x * 128;
  const int col0 = blockIdx.y * 128;

  // A-panel fill coords: 128 rows x 32 K, 16 halfs per thread
  const int a_lrow = tid >> 1;             // 0..127
  const int a_bk   = (tid & 1) * 16;       // 0 or 16
  const int a_row  = row0 + a_lrow;
  const bool a_ok  = !MGUARD || (a_row < M);
  // B-panel fill coords: 32 K x 128 cols, 16 halfs per thread
  const int b_k    = tid >> 3;             // 0..31
  const int b_cb   = (tid & 7) * 16;       // 0..112

  const int lane = tid & 31;
  const int wave = tid >> 5;               // 0..7 -> 16-row strip
  const int half = lane >> 4;
  const int lr   = lane & 15;

  v8f acc[8] = {};

  auto step = [&](int k0, bool guard) {
    // ---- stage A panel ----
    {
      _Float16 va[16];
      if (a_ok && !guard) {
        const float4* src = (const float4*)(A + (size_t)a_row * K + k0 + a_bk);
        #pragma unroll
        for (int q = 0; q < 4; ++q) {
          float4 f = src[q];
          va[q * 4 + 0] = (_Float16)f.x; va[q * 4 + 1] = (_Float16)f.y;
          va[q * 4 + 2] = (_Float16)f.z; va[q * 4 + 3] = (_Float16)f.w;
        }
      } else {
        #pragma unroll
        for (int e = 0; e < 16; ++e) {
          int kk = k0 + a_bk + e;
          float v = (a_ok && kk < K) ? A[(size_t)a_row * K + kk] : 0.0f;
          va[e] = (_Float16)v;
        }
      }
      v8h lo, hi;
      #pragma unroll
      for (int e = 0; e < 8; ++e) { lo[e] = va[e]; hi[e] = va[e + 8]; }
      *(v8h*)&sA[a_lrow * LDS_STRIDE + a_bk]     = lo;
      *(v8h*)&sA[a_lrow * LDS_STRIDE + a_bk + 8] = hi;
    }
    // ---- stage B panel, transposed: sB[col][k] ----
    {
      int kk = k0 + b_k;
      _Float16 vb[16];
      if (!guard) {
        const float4* src = (const float4*)(B + (size_t)kk * N + col0 + b_cb);
        #pragma unroll
        for (int q = 0; q < 4; ++q) {
          float4 f = src[q];
          vb[q * 4 + 0] = (_Float16)f.x; vb[q * 4 + 1] = (_Float16)f.y;
          vb[q * 4 + 2] = (_Float16)f.z; vb[q * 4 + 3] = (_Float16)f.w;
        }
      } else {
        #pragma unroll
        for (int e = 0; e < 16; ++e) {
          float v = (kk < K) ? B[(size_t)kk * N + col0 + b_cb + e] : 0.0f;
          vb[e] = (_Float16)v;
        }
      }
      #pragma unroll
      for (int c = 0; c < 16; ++c) sB[(b_cb + c) * LDS_STRIDE + b_k] = vb[c];
    }
    __syncthreads();

    // ---- fragments + 8 WMMA per wave ----
    // A fragment (16-bit A 16x32): element j -> K = half*8 + j + (j>=8?8:0)
    const _Float16* pa = &sA[(wave * 16 + lr) * LDS_STRIDE + half * 8];
    v8h a0 = *(const v8h*)pa;          // K = half*8 + 0..7
    v8h a1 = *(const v8h*)(pa + 16);   // K = half*8 + 16..23
    v16h afrag = __builtin_shufflevector(a0, a1, 0, 1, 2, 3, 4, 5, 6, 7,
                                         8, 9, 10, 11, 12, 13, 14, 15);
    #pragma unroll
    for (int t = 0; t < 8; ++t) {
      // B fragment (16-bit B 32x16): element j -> K = half*8 + j + (j>=8?8:0), col = lr
      const _Float16* pb = &sB[(t * 16 + lr) * LDS_STRIDE + half * 8];
      v8h b0 = *(const v8h*)pb;
      v8h b1 = *(const v8h*)(pb + 16);
      v16h bfrag = __builtin_shufflevector(b0, b1, 0, 1, 2, 3, 4, 5, 6, 7,
                                           8, 9, 10, 11, 12, 13, 14, 15);
      acc[t] = __builtin_amdgcn_wmma_f32_16x16x32_f16(
          /*neg_a=*/false, afrag, /*neg_b=*/false, bfrag,
          /*c_mod=*/(short)0, acc[t], /*reuse_a=*/false, /*reuse_b=*/false);
    }
    __syncthreads();
  };

  const int kfull = K & ~31;
  for (int k0 = 0; k0 < kfull; k0 += 32) step(k0, false);
  if (kfull < K) step(kfull, true);

  #pragma unroll
  for (int t = 0; t < 8; ++t) {
    int cc = col0 + t * 16 + lr;
    #pragma unroll
    for (int rb = 0; rb < 8; ++rb) {
      int rr = row0 + wave * 16 + rb + half * 8;
      if (!MGUARD || rr < M) {
        float v = acc[t][rb];
        if (relu) v = fmaxf(v, 0.0f);
        C[(size_t)rr * N + cc] = v;
      }
    }
  }
}

// ---------------------------------------------------------------------------
// Per-column mean and 1/sqrt(var+eps) for batchnorm. One block per column.
// ---------------------------------------------------------------------------
__global__ void k_colstats(const float* __restrict__ h, int n, int c,
                           float* __restrict__ mu, float* __restrict__ rstd)
{
  const int col = blockIdx.x;
  const int tid = threadIdx.x;
  float s = 0.0f, s2 = 0.0f;
  for (int i = tid; i < n; i += blockDim.x) {
    float v = h[(size_t)i * c + col];
    s += v; s2 += v * v;
  }
  __shared__ float rs[256], rs2[256];
  rs[tid] = s; rs2[tid] = s2;
  __syncthreads();
  for (int st = 128; st > 0; st >>= 1) {
    if (tid < st) { rs[tid] += rs[tid + st]; rs2[tid] += rs2[tid + st]; }
    __syncthreads();
  }
  if (tid == 0) {
    float m   = rs[0] / (float)n;
    float var = rs2[0] / (float)n - m * m;
    mu[col]   = m;
    rstd[col] = rsqrtf(var + EPSV);
  }
}

// out = relu(gamma*(hw-mu)*rstd + beta) [+ sc]
__global__ void k_bn_relu_add(const float* __restrict__ hw,
                              const float* __restrict__ mu, const float* __restrict__ rstd,
                              const float* __restrict__ gamma, const float* __restrict__ beta,
                              const float* __restrict__ sc, float* __restrict__ out,
                              int n, int c)
{
  size_t idx = (size_t)blockIdx.x * blockDim.x + threadIdx.x;
  if (idx < (size_t)n * c) {
    int col = (int)(idx % c);
    float v = gamma[col] * (hw[idx] - mu[col]) * rstd[col] + beta[col];
    v = fmaxf(v, 0.0f);
    if (sc) v += sc[idx];
    out[idx] = v;
  }
}

// Pair pooling of features: h2[i] = 0.5*(h[2i]+h[2i+1])
__global__ void k_pool_feat(const float* __restrict__ h, float* __restrict__ h2,
                            int n2, int c)
{
  size_t idx = (size_t)blockIdx.x * blockDim.x + threadIdx.x;
  if (idx < (size_t)n2 * c) {
    int i = (int)(idx / c), ch = (int)(idx % c);
    h2[idx] = 0.5f * (h[(size_t)(2 * i) * c + ch] + h[(size_t)(2 * i + 1) * c + ch]);
  }
}

// Pair pooling of geometry
__global__ void k_pool_geo(const float* __restrict__ pos, const int* __restrict__ seq,
                           const float* __restrict__ ori, const int* __restrict__ batch,
                           float* __restrict__ pos2, int* __restrict__ seq2,
                           float* __restrict__ ori2, int* __restrict__ batch2, int n2)
{
  int i = blockIdx.x * blockDim.x + threadIdx.x;
  if (i < n2) {
    int a = 2 * i, b = 2 * i + 1;
    pos2[i * 3 + 0] = 0.5f * (pos[a * 3 + 0] + pos[b * 3 + 0]);
    pos2[i * 3 + 1] = 0.5f * (pos[a * 3 + 1] + pos[b * 3 + 1]);
    pos2[i * 3 + 2] = 0.5f * (pos[a * 3 + 2] + pos[b * 3 + 2]);
    float oxm = 0.5f * (ori[a * 3 + 0] + ori[b * 3 + 0]);
    float oym = 0.5f * (ori[a * 3 + 1] + ori[b * 3 + 1]);
    float ozm = 0.5f * (ori[a * 3 + 2] + ori[b * 3 + 2]);
    float nrm = fmaxf(sqrtf(oxm * oxm + oym * oym + ozm * ozm), 1e-8f);
    ori2[i * 3 + 0] = oxm / nrm; ori2[i * 3 + 1] = oym / nrm; ori2[i * 3 + 2] = ozm / nrm;
    seq2[i]   = seq[a] / 2;
    batch2[i] = batch[a];
  }
}

// Segment mean over contiguous equal-sized segments
__global__ void k_segmean(const float* __restrict__ h, float* __restrict__ out,
                          int rows_per, int c)
{
  int b   = blockIdx.x;
  int col = blockIdx.y * blockDim.x + threadIdx.x;
  if (col < c) {
    float s = 0.0f;
    for (int r = 0; r < rows_per; ++r)
      s += h[((size_t)b * rows_per + r) * c + col];
    out[(size_t)b * c + col] = s / (float)rows_per;
  }
}

// Copy [rows, cin] into columns [coff, coff+cin) of [rows, ctot]
__global__ void k_copycols(const float* __restrict__ in, float* __restrict__ out,
                           int rows, int cin, int ctot, int coff)
{
  size_t idx = (size_t)blockIdx.x * blockDim.x + threadIdx.x;
  if (idx < (size_t)rows * cin) {
    int r = (int)(idx / cin), c = (int)(idx % cin);
    out[(size_t)r * ctot + coff + c] = in[idx];
  }
}

// Final select: out[b] = (sum(domain[b]) != 0) ? out_com[b] : out_inc[b]
__global__ void k_select(const float* __restrict__ domain,
                         const float* __restrict__ ocom, const float* __restrict__ oinc,
                         float* __restrict__ out, int dom, int ncls)
{
  int b = blockIdx.x, tid = threadIdx.x;
  __shared__ float red[128];
  float s = 0.0f;
  for (int j = tid; j < dom; j += blockDim.x) s += domain[(size_t)b * dom + j];
  red[tid] = s;
  __syncthreads();
  for (int st = 64; st > 0; st >>= 1) {
    if (tid < st) red[tid] += red[tid + st];
    __syncthreads();
  }
  bool has = red[0] != 0.0f;
  for (int j = tid; j < ncls; j += blockDim.x)
    out[(size_t)b * ncls + j] = has ? ocom[(size_t)b * ncls + j]
                                    : oinc[(size_t)b * ncls + j];
}

// ---------------------------------------------------------------------------
// Host orchestration
// ---------------------------------------------------------------------------
extern "C" void kernel_launch(void* const* d_in, const int* in_sizes, int n_in,
                              void* d_out, int out_size, void* d_ws, size_t ws_size,
                              hipStream_t stream)
{
  (void)in_sizes; (void)n_in; (void)out_size; (void)ws_size;
  const int N0 = 32 * 1024, Bn = 32, NCLS = 384, PROJ = 1280, DOM = 256;

  auto F = [&](int i) -> const float* { return (const float*)d_in[i]; };
  auto I = [&](int i) -> const int*   { return (const int*)d_in[i]; };

  const int*   x       = I(0);
  const float* pos     = F(1);
  const int*   seq     = I(2);
  const float* ori     = F(3);
  const float* domain  = F(4);
  const float* seq_emb = F(5);
  const int*   batch   = I(6);

  // params: jax tree-flatten order (alphabetical keys per dict level)
  struct BlockP {
    const float *bk1, *bk2, *beta, *gamma, *wk1, *wk2, *wlin, *wsc;
  } blk[8];
  const int   ics[8]   = {16, 128, 128, 256, 256, 512, 512, 1024};
  const int   couts[8] = {128, 128, 256, 256, 512, 512, 1024, 1024};
  const float rad[8]   = {4.f, 4.f, 8.f, 8.f, 12.f, 12.f, 16.f, 16.f};
  int p = 7;
  for (int i = 0; i < 8; ++i) {
    bool has_wsc = (ics[i] != couts[i]);
    blk[i].bk1 = F(p + 0); blk[i].bk2 = F(p + 1);
    blk[i].beta = F(p + 2); blk[i].gamma = F(p + 3);
    blk[i].wk1 = F(p + 4); blk[i].wk2 = F(p + 5); blk[i].wlin = F(p + 6);
    blk[i].wsc = has_wsc ? F(p + 7) : nullptr;
    p += has_wsc ? 8 : 7;
  }
  const float *cls1_beta = F(p++), *cls1_gamma = F(p++), *cls1_w1 = F(p++), *cls1_w2 = F(p++);
  const float *cls2_beta = F(p++), *cls2_gamma = F(p++), *cls2_w1 = F(p++), *cls2_w2 = F(p++);
  const float *complete_w1 = F(p++), *complete_w2 = F(p++);
  const float *embT = F(p++);
  const float *pd_w1 = F(p++), *pd_w2 = F(p++);
  const float *psq_w1 = F(p++), *psq_w2 = F(p++);
  const float *pst_w1 = F(p++), *pst_w2 = F(p++);
  const float *pss_w1 = F(p++), *pss_w2 = F(p++);

  char* wsb = (char*)d_ws;
  size_t off = 0;
  auto alloc = [&](size_t bytes) -> void* {
    void* q = wsb + off;
    off = (off + bytes + 255) & ~(size_t)255;
    return q;
  };
  const size_t BIG = (size_t)N0 * 128 * sizeof(float);
  float* hA    = (float*)alloc(BIG);
  float* hB    = (float*)alloc(BIG);
  float* hPre  = (float*)alloc(BIG);
  float* hHW   = (float*)alloc(BIG);
  float* hSC   = (float*)alloc(BIG);
  float* posA  = (float*)alloc((size_t)(N0 / 2) * 3 * sizeof(float));
  float* posB  = (float*)alloc((size_t)(N0 / 2) * 3 * sizeof(float));
  float* oriA  = (float*)alloc((size_t)(N0 / 2) * 3 * sizeof(float));
  float* oriB  = (float*)alloc((size_t)(N0 / 2) * 3 * sizeof(float));
  int*   seqA  = (int*)alloc((size_t)(N0 / 2) * sizeof(int));
  int*   seqB  = (int*)alloc((size_t)(N0 / 2) * sizeof(int));
  int*   batA  = (int*)alloc((size_t)(N0 / 2) * sizeof(int));
  int*   batB  = (int*)alloc((size_t)(N0 / 2) * sizeof(int));
  float* muBuf = (float*)alloc(1280 * sizeof(float));
  float* rsBuf = (float*)alloc(1280 * sizeof(float));
  float* hG    = (float*)alloc((size_t)Bn * 1024 * sizeof(float));
  float* t1    = (float*)alloc((size_t)Bn * PROJ * sizeof(float));
  float* sfeat = (float*)alloc((size_t)Bn * PROJ * sizeof(float));
  float* qfeat = (float*)alloc((size_t)Bn * PROJ * sizeof(float));
  float* dfeat = (float*)alloc((size_t)Bn * PROJ * sizeof(float));
  float* cat2  = (float*)alloc((size_t)Bn * 2 * PROJ * sizeof(float));
  float* cat3  = (float*)alloc((size_t)Bn * 3 * PROJ * sizeof(float));
  float* fInc  = (float*)alloc((size_t)Bn * PROJ * sizeof(float));
  float* fCom  = (float*)alloc((size_t)Bn * PROJ * sizeof(float));
  float* tMid  = (float*)alloc((size_t)Bn * PROJ * sizeof(float));
  float* oInc  = (float*)alloc((size_t)Bn * NCLS * sizeof(float));
  float* oCom  = (float*)alloc((size_t)Bn * NCLS * sizeof(float));

  auto gemm = [&](const float* A, const float* W, float* C,
                  int M, int K, int Nc, int relu) {
    dim3 g((M + 127) / 128, (Nc + 127) / 128);
    if (M % 128 == 0)
      k_gemm_wmma<false><<<g, dim3(256), 0, stream>>>(A, W, C, M, K, Nc, relu);
    else
      k_gemm_wmma<true><<<g, dim3(256), 0, stream>>>(A, W, C, M, K, Nc, relu);
  };

  // ---- embedding ----
  k_embed<<<(N0 * 16 + 255) / 256, 256, 0, stream>>>(x, embT, hA, N0);

  // ---- graph conv blocks ----
  int n = N0;
  float* curH = hA;
  const float* curPos = pos; const int* curSeq = seq;
  const float* curOri = ori; const int* curBat = batch;
  int geoPing = 0;

  for (int i = 0; i < 8; ++i) {
    const int ic = ics[i], cout = couts[i];
    k_msg<<<n, 256, 0, stream>>>(curH, curPos, curSeq, curOri, curBat,
                                 blk[i].wk1, blk[i].bk1, blk[i].wk2, blk[i].bk2,
                                 hPre, n, ic, rad[i]);
    gemm(hPre, blk[i].wlin, hHW, n, ic, cout, 0);
    const float* scPtr = curH;
    if (blk[i].wsc) { gemm(curH, blk[i].wsc, hSC, n, ic, cout, 0); scPtr = hSC; }
    k_colstats<<<cout, 256, 0, stream>>>(hHW, n, cout, muBuf, rsBuf);
    float* nextH = (curH == hA) ? hB : hA;
    size_t tot = (size_t)n * cout;
    k_bn_relu_add<<<(int)((tot + 255) / 256), 256, 0, stream>>>(
        hHW, muBuf, rsBuf, blk[i].gamma, blk[i].beta, scPtr, nextH, n, cout);
    curH = nextH;

    if (i == 7) {
      k_segmean<<<dim3(Bn, (cout + 255) / 256), 256, 0, stream>>>(
          curH, hG, n / Bn, cout);
    } else if (i % 2 == 1) {
      int n2 = n / 2;
      float* pooled = (curH == hA) ? hB : hA;
      k_pool_feat<<<(int)(((size_t)n2 * cout + 255) / 256), 256, 0, stream>>>(
          curH, pooled, n2, cout);
      float* pOut = geoPing ? posB : posA;  float* oOut = geoPing ? oriB : oriA;
      int*   sOut = geoPing ? seqB : seqA;  int*   bOut = geoPing ? batB : batA;
      k_pool_geo<<<(n2 + 255) / 256, 256, 0, stream>>>(
          curPos, curSeq, curOri, curBat, pOut, sOut, oOut, bOut, n2);
      curH = pooled; n = n2;
      curPos = pOut; curOri = oOut; curSeq = sOut; curBat = bOut;
      geoPing ^= 1;
    }
  }

  // ---- projectors: f = relu(h @ w1) @ w2 ----
  gemm(hG, pst_w1, t1, Bn, 1024, PROJ, 1);
  gemm(t1, pst_w2, sfeat, Bn, PROJ, PROJ, 0);
  gemm(seq_emb, psq_w1, t1, Bn, 1280, PROJ, 1);
  gemm(t1, psq_w2, qfeat, Bn, PROJ, PROJ, 0);
  gemm(domain, pd_w1, t1, Bn, DOM, PROJ, 1);
  gemm(t1, pd_w2, dfeat, Bn, PROJ, PROJ, 0);

  // ---- concatenations ----
  int cpThreads = (Bn * PROJ + 255) / 256;
  k_copycols<<<cpThreads, 256, 0, stream>>>(sfeat, cat2, Bn, PROJ, 2 * PROJ, 0);
  k_copycols<<<cpThreads, 256, 0, stream>>>(qfeat, cat2, Bn, PROJ, 2 * PROJ, PROJ);
  k_copycols<<<cpThreads, 256, 0, stream>>>(sfeat, cat3, Bn, PROJ, 3 * PROJ, 0);
  k_copycols<<<cpThreads, 256, 0, stream>>>(qfeat, cat3, Bn, PROJ, 3 * PROJ, PROJ);
  k_copycols<<<cpThreads, 256, 0, stream>>>(dfeat, cat3, Bn, PROJ, 3 * PROJ, 2 * PROJ);

  // ---- struct_seq / complete projectors ----
  gemm(cat2, pss_w1, t1, Bn, 2 * PROJ, PROJ, 1);
  gemm(t1, pss_w2, fInc, Bn, PROJ, PROJ, 0);
  gemm(cat3, complete_w1, t1, Bn, 3 * PROJ, PROJ, 1);
  gemm(t1, complete_w2, fCom, Bn, PROJ, PROJ, 0);

  // ---- classifier MLPs: relu(bn(h@w1)) @ w2 ----
  gemm(fInc, cls2_w1, tMid, Bn, PROJ, PROJ, 0);
  k_colstats<<<PROJ, 256, 0, stream>>>(tMid, Bn, PROJ, muBuf, rsBuf);
  k_bn_relu_add<<<(Bn * PROJ + 255) / 256, 256, 0, stream>>>(
      tMid, muBuf, rsBuf, cls2_gamma, cls2_beta, nullptr, t1, Bn, PROJ);
  gemm(t1, cls2_w2, oInc, Bn, PROJ, NCLS, 0);

  gemm(fCom, cls1_w1, tMid, Bn, PROJ, PROJ, 0);
  k_colstats<<<PROJ, 256, 0, stream>>>(tMid, Bn, PROJ, muBuf, rsBuf);
  k_bn_relu_add<<<(Bn * PROJ + 255) / 256, 256, 0, stream>>>(
      tMid, muBuf, rsBuf, cls1_gamma, cls1_beta, nullptr, t1, Bn, PROJ);
  gemm(t1, cls1_w2, oCom, Bn, PROJ, NCLS, 0);

  // ---- final per-batch select ----
  k_select<<<Bn, 128, 0, stream>>>(domain, oCom, oInc, (float*)d_out, DOM, NCLS);
}
